// LayoutGraphModel_86663850098883
// MI455X (gfx1250) — matrix operand order repaired and checked by reading, not compiled, not run on hardware
//
#include <hip/hip_runtime.h>

// ---------------- model constants (match reference) ----------------
#define NFEAT   126
#define OPD     64
#define INDP    192     // 190 padded to mult of 32
#define IND_R   190
#define HIDP    384     // 380 padded
#define HID_R   380
#define GIN     256
#define GH      512
#define NEGSL   0.01f
#define LDSPAD  8       // bf16 elements of per-row padding in LDS B tiles

typedef __attribute__((ext_vector_type(16))) __bf16 v16bf;
typedef __attribute__((ext_vector_type(8)))  __bf16 v8bf;
typedef __attribute__((ext_vector_type(4)))  __bf16 v4bf;
typedef __attribute__((ext_vector_type(8)))  float  v8f;

extern __shared__ __bf16 smem[];   // dynamic LDS: B tile(s), row stride K+LDSPAD

// ---------------- CDNA5 async global->LDS copy (ASYNCcnt-tracked) -------
__device__ __forceinline__ void async_ld_b128(unsigned lds_off, const __bf16* g) {
    asm volatile("global_load_async_to_lds_b128 %0, %1, off"
                 :: "v"(lds_off), "v"((unsigned long long)(uintptr_t)g)
                 : "memory");
}
__device__ __forceinline__ void wait_async0() {
#if __has_builtin(__builtin_amdgcn_s_wait_asynccnt)
    __builtin_amdgcn_s_wait_asynccnt(0);
#else
    asm volatile("s_wait_asynccnt 0x0" ::: "memory");
#endif
}

// ---------------- WMMA fragment loaders (CDNA5 bf16 16x16x32 layouts) ----
// A (16x32): lane L -> row m = L&15; half = L>>4.
//   halfs e=0..7  at row[ half*8 + e ];  halfs e=8..15 at row[ 16 + half*8 + (e-8) ]
__device__ __forceinline__ v16bf load_a_frag(const __bf16* __restrict__ rowk, int half) {
    v8bf lo = *(const v8bf*)(rowk + half * 8);
    v8bf hi = *(const v8bf*)(rowk + 16 + half * 8);
    v16bf r;
#pragma unroll
    for (int i = 0; i < 8; ++i) { r[i] = lo[i]; r[i + 8] = hi[i]; }
    return r;
}
// B (32x16) from row-major Bt[n, k]: lane L -> col n = L&15; half = L>>4;
//   halfs e=0..7  at colrow[ half*8 + e ];  halfs e=8..15 at colrow[ 16 + half*8 + (e-8) ]
__device__ __forceinline__ v16bf load_b_frag(const __bf16* colk, int half) {
    v8bf lo = *(const v8bf*)(colk + half * 8);
    v8bf hi = *(const v8bf*)(colk + 16 + half * 8);
    v16bf r;
#pragma unroll
    for (int i = 0; i < 8; ++i) { r[i] = lo[i]; r[i + 8] = hi[i]; }
    return r;
}

#define WMMA_BF16(a, b, c) \
    __builtin_amdgcn_wmma_f32_16x16x32_bf16(false, (a), false, (b), (short)0, (c), false, false)

// one K-chain: A from global, B tile from LDS at sbase. Both A and the 4 B
// fragments are software-pipelined one K-step ahead so their loads retire
// under the previous group's 4 back-to-back WMMAs (partial DScnt waits).
// Last K-step peeled -> branch-free loop body (lets the unroller rotate the
// ping-pong registers without v_mov copies).
__device__ __forceinline__ void wmma_chain_lds(const __bf16* __restrict__ A,
                                               size_t arow, int sbase, int K, int K8,
                                               int l16, int half, v8f acc[4]) {
    const __bf16* __restrict__ ap = A + arow;
    const int sb0 = sbase + l16 * K8;
    v16bf a  = load_a_frag(ap, half);
    v16bf b0 = load_b_frag(&smem[sb0],           half);
    v16bf b1 = load_b_frag(&smem[sb0 + 16 * K8], half);
    v16bf b2 = load_b_frag(&smem[sb0 + 32 * K8], half);
    v16bf b3 = load_b_frag(&smem[sb0 + 48 * K8], half);
    for (int k0 = 32; k0 < K; k0 += 32) {
        v16bf an = load_a_frag(ap + k0, half);
        v16bf n0 = load_b_frag(&smem[sb0 + k0],           half);
        v16bf n1 = load_b_frag(&smem[sb0 + k0 + 16 * K8], half);
        v16bf n2 = load_b_frag(&smem[sb0 + k0 + 32 * K8], half);
        v16bf n3 = load_b_frag(&smem[sb0 + k0 + 48 * K8], half);
        acc[0] = WMMA_BF16(a, b0, acc[0]);
        acc[1] = WMMA_BF16(a, b1, acc[1]);
        acc[2] = WMMA_BF16(a, b2, acc[2]);
        acc[3] = WMMA_BF16(a, b3, acc[3]);
        a = an; b0 = n0; b1 = n1; b2 = n2; b3 = n3;
    }
    acc[0] = WMMA_BF16(a, b0, acc[0]);
    acc[1] = WMMA_BF16(a, b1, acc[1]);
    acc[2] = WMMA_BF16(a, b2, acc[2]);
    acc[3] = WMMA_BF16(a, b3, acc[3]);
}

// ---------------- fused GEMM: C = A@B (+ A2@B2) + bias, optional ReLU ----
// A:[M,K] bf16 row-major; Bt:[Npad,K] bf16 (B transposed). Block = 8 waves,
// each wave owns a 16x64 tile; B tile staged in LDS via async copies.
__global__ void gemm_wmma_kernel(const __bf16* __restrict__ A,
                                 const __bf16* __restrict__ Bt,
                                 const __bf16* __restrict__ A2,
                                 const __bf16* __restrict__ Bt2,
                                 const float* __restrict__ bias,
                                 float* __restrict__ Cf,
                                 __bf16* __restrict__ Cb,
                                 int M, int K, int Npad, int Nreal, int act) {
    const int K8 = K + LDSPAD;
    const int n0 = blockIdx.y * 64;

    const int wave = threadIdx.x >> 5;
    const int lane = threadIdx.x & 31;
    const int half = lane >> 4;
    const int l16  = lane & 15;
    const int mt   = blockIdx.x * (blockDim.x >> 5) + wave;
    const bool active = (mt * 16 < M);
    const size_t arow = (size_t)(mt * 16 + l16) * K;

    // ---- stage B tile(s): 64 rows x K bf16, async b128 per lane ----
    const int kchunks = K >> 3;               // 16B chunks per row
    const int total   = 64 * kchunks;
    for (int c = threadIdx.x; c < total; c += blockDim.x) {
        const int row = c / kchunks;
        const int kc  = (c - row * kchunks) << 3;
        async_ld_b128((unsigned)(uintptr_t)&smem[row * K8 + kc],
                      Bt + (size_t)(n0 + row) * K + kc);
        if (A2)
            async_ld_b128((unsigned)(uintptr_t)&smem[(64 + row) * K8 + kc],
                          Bt2 + (size_t)(n0 + row) * K + kc);
    }
    // warm the A stream while B staging is in flight
    if (active) {
        __builtin_prefetch(A + arow, 0, 1);
        if (A2) __builtin_prefetch(A2 + arow, 0, 1);
    }
    wait_async0();
    __syncthreads();
    if (!active) return;                      // no further barriers below

    v8f acc[4] = {{}, {}, {}, {}};
    wmma_chain_lds(A, arow, 0, K, K8, l16, half, acc);
    if (A2) wmma_chain_lds(A2, arow, 64 * K8, K, K8, l16, half, acc);

    const int mrow = mt * 16;
#pragma unroll
    for (int j = 0; j < 4; ++j) {
        const int col = n0 + j * 16 + l16;
        const float bv = (bias && col < Nreal) ? bias[col] : 0.f;
#pragma unroll
        for (int r = 0; r < 8; ++r) {
            float v = acc[j][r] + bv;
            if (act == 1) v = fmaxf(v, 0.f);
            const size_t idx = (size_t)(mrow + half * 8 + r) * Npad + col;
            if (Cf) Cf[idx] = v;
            if (Cb) Cb[idx] = (__bf16)v;
        }
    }
}

// ---------------- small helper kernels ----------------
__global__ void zero_kernel(float* __restrict__ p, size_t n) {
    size_t i = (size_t)blockIdx.x * blockDim.x + threadIdx.x;
    if (i < n) p[i] = 0.f;
}

// W[K,Nout] f32 -> Wt[NoutPad,Kpad] bf16 (transposed, zero padded)
__global__ void wconv_kernel(const float* __restrict__ W, __bf16* __restrict__ Wt,
                             int K, int Nout, int Kpad, int NoutPad) {
    size_t i = (size_t)blockIdx.x * blockDim.x + threadIdx.x;
    if (i >= (size_t)NoutPad * Kpad) return;
    int n = (int)(i / Kpad), k = (int)(i % Kpad);
    float v = (n < Nout && k < K) ? W[(size_t)k * Nout + n] : 0.f;
    Wt[i] = (__bf16)v;
}

// x0[n,c] = concat(node_features, emb[node_ops]) -> bf16 [N, INDP]
__global__ void embed_kernel(const float* __restrict__ nf, const int* __restrict__ ops,
                             const float* __restrict__ emb, __bf16* __restrict__ x0, int N) {
    size_t i = (size_t)blockIdx.x * blockDim.x + threadIdx.x;
    if (i >= (size_t)N * INDP) return;
    int n = (int)(i / INDP), c = (int)(i % INDP);
    float v = 0.f;
    if (c < NFEAT)       v = nf[(size_t)n * NFEAT + c];
    else if (c < IND_R)  v = emb[(size_t)ops[n] * OPD + (c - NFEAT)];
    x0[i] = (__bf16)v;
}

__global__ void deg_kernel(const int* __restrict__ edges, float* __restrict__ deg, int E) {
    int e = blockIdx.x * blockDim.x + threadIdx.x;
    if (e >= E) return;
    atomicAdd(deg + edges[E + e], 1.f);
}

// mean-aggregation scatter: 4 channels / thread, f32 atomics
__global__ void agg_scatter_kernel(const int* __restrict__ edges, const __bf16* __restrict__ X,
                                   float* __restrict__ AGG, int E, int F) {
    const int F4 = F >> 2;
    size_t i = (size_t)blockIdx.x * blockDim.x + threadIdx.x;
    if (i >= (size_t)E * F4) return;
    int e = (int)(i / F4), c = (int)(i % F4) << 2;
    int src = edges[e], dst = edges[E + e];
    v4bf x = *(const v4bf*)(X + (size_t)src * F + c);
    float* base = AGG + (size_t)dst * F + c;
    atomicAdd(base + 0, (float)x[0]);
    atomicAdd(base + 1, (float)x[1]);
    atomicAdd(base + 2, (float)x[2]);
    atomicAdd(base + 3, (float)x[3]);
}

__global__ void agg_fin_kernel(const float* __restrict__ AGG, const float* __restrict__ deg,
                               __bf16* __restrict__ AB, int N, int F) {
    size_t i = (size_t)blockIdx.x * blockDim.x + threadIdx.x;
    if (i >= (size_t)N * F) return;
    int n = (int)(i / F);
    AB[i] = (__bf16)(AGG[i] / fmaxf(deg[n], 1.f));
}

__device__ __forceinline__ float block_reduce_sum(float v, float* red) {
#pragma unroll
    for (int off = 16; off > 0; off >>= 1) v += __shfl_down(v, off);
    int wid = threadIdx.x >> 5, lane = threadIdx.x & 31;
    if (lane == 0) red[wid] = v;
    __syncthreads();
    float t = (threadIdx.x < (blockDim.x >> 5)) ? red[threadIdx.x] : 0.f;
#pragma unroll
    for (int off = 4; off > 0; off >>= 1) t += __shfl_down(t, off);
    if (threadIdx.x == 0) red[0] = t;
    __syncthreads();
    return red[0];
}

// per-row L2 normalize + leaky relu, f32 -> bf16
__global__ void norm_leaky_kernel(const float* __restrict__ H, __bf16* __restrict__ X, int F) {
    __shared__ float red[8];
    const int n = blockIdx.x;
    float s = 0.f;
    for (int c = threadIdx.x; c < F; c += blockDim.x) {
        float v = H[(size_t)n * F + c];
        s += v * v;
    }
    float tot = block_reduce_sum(s, red);
    float inv = 1.f / fmaxf(sqrtf(tot), 1e-12f);
    for (int c = threadIdx.x; c < F; c += blockDim.x) {
        float v = H[(size_t)n * F + c] * inv;
        v = (v > 0.f) ? v : NEGSL * v;
        X[(size_t)n * F + c] = (__bf16)v;
    }
}

// ragged segment sum over contiguous node ranges
__global__ void seg_sum_kernel(const float* __restrict__ H, const int* __restrict__ sep,
                               float* __restrict__ SS, int F) {
    int s = blockIdx.x;
    int c = blockIdx.y * blockDim.x + threadIdx.x;
    if (c >= F) return;
    int start = (s == 0) ? 0 : sep[s - 1];
    int end   = sep[s];
    float acc = 0.f;
    for (int n = start; n < end; ++n) acc += H[(size_t)n * F + c];
    SS[(size_t)s * F + c] = acc;
}

__global__ void final_kernel(const float* __restrict__ SS, const float* __restrict__ Wf,
                             const float* __restrict__ bf_, float* __restrict__ out, int F) {
    __shared__ float red[8];
    const int s = blockIdx.x;
    float acc = 0.f;
    for (int c = threadIdx.x; c < F; c += blockDim.x)
        acc += SS[(size_t)s * F + c] * Wf[c];
    float tot = block_reduce_sum(acc, red);
    if (threadIdx.x == 0) out[s] = tot + bf_[0];
}

// ---------------- host orchestration ----------------
extern "C" void kernel_launch(void* const* d_in, const int* in_sizes, int n_in,
                              void* d_out, int out_size, void* d_ws, size_t ws_size,
                              hipStream_t stream) {
    const float* node_features = (const float*)d_in[0];
    const int*   node_ops      = (const int*)d_in[1];
    const int*   edges         = (const int*)d_in[2];
    const int*   sep           = (const int*)d_in[3];
    /* d_in[4] = batches (identity ordering: sorted, C per graph) */
    const float* emb = (const float*)d_in[5];
    const float* W1  = (const float*)d_in[6];  const float* b1  = (const float*)d_in[7];
    const float* W2  = (const float*)d_in[8];  const float* b2  = (const float*)d_in[9];
    const float* Wl0 = (const float*)d_in[10]; const float* bl0 = (const float*)d_in[11];
    const float* Wr0 = (const float*)d_in[12];
    const float* Wl1 = (const float*)d_in[13]; const float* bl1 = (const float*)d_in[14];
    const float* Wr1 = (const float*)d_in[15];
    const float* Wl2 = (const float*)d_in[16]; const float* bl2 = (const float*)d_in[17];
    const float* Wr2 = (const float*)d_in[18];
    const float* Wf  = (const float*)d_in[19]; const float* bfp = (const float*)d_in[20];

    const int N = in_sizes[1];       // 50000 (multiple of 16)
    const int E = in_sizes[2] / 2;   // 160000
    const int S = in_sizes[3];       // 80

    // ---- workspace carving (256B aligned); BIG region is time-shared ----
    char* ws = (char*)d_ws;
    size_t off = 0;
    auto carve = [&](size_t bytes) -> char* {
        off = (off + 255) & ~(size_t)255;
        char* p = ws + off; off += bytes; return p;
    };
    char*   big   = carve((size_t)N * GH * 4);     // aggf / hf / (x0 + y1)
    __bf16* xbf   = (__bf16*)carve((size_t)N * GH * 2);
    __bf16* aggb  = (__bf16*)carve((size_t)N * GH * 2);
    float*  deg   = (float*)carve((size_t)N * 4);
    float*  segs  = (float*)carve((size_t)S * GH * 4);
    __bf16* w1t   = (__bf16*)carve((size_t)HIDP * INDP * 2);
    __bf16* w2t   = (__bf16*)carve((size_t)GIN * HIDP * 2);
    __bf16* wl0t  = (__bf16*)carve((size_t)GH * GIN * 2);
    __bf16* wr0t  = (__bf16*)carve((size_t)GH * GIN * 2);
    __bf16* wl1t  = (__bf16*)carve((size_t)GH * GH * 2);
    __bf16* wr1t  = (__bf16*)carve((size_t)GH * GH * 2);
    __bf16* wl2t  = (__bf16*)carve((size_t)GH * GH * 2);
    __bf16* wr2t  = (__bf16*)carve((size_t)GH * GH * 2);
    (void)ws_size; (void)n_in; (void)out_size;

    float*  aggf = (float*)big;   // aliases: live ranges are disjoint
    float*  hf   = (float*)big;
    __bf16* x0   = (__bf16*)big;
    __bf16* y1   = (__bf16*)(big + (((size_t)N * INDP * 2 + 255) & ~(size_t)255));

    const dim3 blk(256);
    const int  mtiles = N / 16;
    const unsigned gx = (unsigned)((mtiles + 7) / 8);
    auto flat = [](size_t n) { return dim3((unsigned)((n + 255) / 256)); };
    auto ldsB = [](int K, bool dual) -> size_t {
        return (size_t)(dual ? 2 : 1) * 64 * (K + LDSPAD) * 2;
    };

    // ---- weight conversion (f32 -> transposed padded bf16) ----
    wconv_kernel<<<flat((size_t)HIDP * INDP), blk, 0, stream>>>(W1, w1t, IND_R, HID_R, INDP, HIDP);
    wconv_kernel<<<flat((size_t)GIN * HIDP), blk, 0, stream>>>(W2, w2t, HID_R, GIN, HIDP, GIN);
    wconv_kernel<<<flat((size_t)GH * GIN),  blk, 0, stream>>>(Wl0, wl0t, GIN, GH, GIN, GH);
    wconv_kernel<<<flat((size_t)GH * GIN),  blk, 0, stream>>>(Wr0, wr0t, GIN, GH, GIN, GH);
    wconv_kernel<<<flat((size_t)GH * GH),   blk, 0, stream>>>(Wl1, wl1t, GH, GH, GH, GH);
    wconv_kernel<<<flat((size_t)GH * GH),   blk, 0, stream>>>(Wr1, wr1t, GH, GH, GH, GH);
    wconv_kernel<<<flat((size_t)GH * GH),   blk, 0, stream>>>(Wl2, wl2t, GH, GH, GH, GH);
    wconv_kernel<<<flat((size_t)GH * GH),   blk, 0, stream>>>(Wr2, wr2t, GH, GH, GH, GH);

    // ---- node MLP ----
    embed_kernel<<<flat((size_t)N * INDP), blk, 0, stream>>>(node_features, node_ops, emb, x0, N);
    gemm_wmma_kernel<<<dim3(gx, HIDP / 64), blk, ldsB(INDP, false), stream>>>(
        x0, w1t, nullptr, nullptr, b1, nullptr, y1, N, INDP, HIDP, HID_R, /*relu*/1);
    gemm_wmma_kernel<<<dim3(gx, GIN / 64), blk, ldsB(HIDP, false), stream>>>(
        y1, w2t, nullptr, nullptr, b2, nullptr, xbf, N, HIDP, GIN, GIN, 0);

    // ---- degrees (once) ----
    zero_kernel<<<flat((size_t)N), blk, 0, stream>>>(deg, (size_t)N);
    deg_kernel<<<flat((size_t)E), blk, 0, stream>>>(edges, deg, E);

    // ---- 3 SAGE layers ----
    const __bf16* wls[3] = {wl0t, wl1t, wl2t};
    const __bf16* wrs[3] = {wr0t, wr1t, wr2t};
    const float*  bls[3] = {bl0, bl1, bl2};
    const int     fin_[3] = {GIN, GH, GH};
    for (int layer = 0; layer < 3; ++layer) {
        const int F = fin_[layer];
        zero_kernel<<<flat((size_t)N * F), blk, 0, stream>>>(aggf, (size_t)N * F);
        agg_scatter_kernel<<<flat((size_t)E * (F >> 2)), blk, 0, stream>>>(edges, xbf, aggf, E, F);
        agg_fin_kernel<<<flat((size_t)N * F), blk, 0, stream>>>(aggf, deg, aggb, N, F);
        gemm_wmma_kernel<<<dim3(gx, GH / 64), blk, ldsB(F, true), stream>>>(
            aggb, wls[layer], xbf, wrs[layer], bls[layer], hf, nullptr, N, F, GH, GH, 0);
        if (layer < 2)
            norm_leaky_kernel<<<N, blk, 0, stream>>>(hf, xbf, GH);
    }

    // ---- readout ----
    seg_sum_kernel<<<dim3((unsigned)S, GH / 256), blk, 0, stream>>>(hf, sep, segs, GH);
    final_kernel<<<S, blk, 0, stream>>>(segs, Wf, bfp, (float*)d_out, GH);
}
